// SageRawSubGraph_90692529422802
// MI455X (gfx1250) — matrix operand order repaired and checked by reading, not compiled, not run on hardware
//
#include <hip/hip_runtime.h>

#define N_NODES 4096
#define N_EDGES 65536
#define LEAKY   0.01f
#define BN_EPS  1e-5f

// ---------------- WMMA types ----------------
typedef __attribute__((ext_vector_type(16))) __bf16 bf16x16;
typedef __attribute__((ext_vector_type(8)))  float  floatx8;

union Frag16 { bf16x16 v; unsigned int u[8]; };

// cheap f32 -> bf16 pack (round-half-up)
__device__ __forceinline__ unsigned int pack_bf16x2(float x, float y) {
  unsigned int lo = (__float_as_uint(x) + 0x8000u) >> 16;
  unsigned int hi = (__float_as_uint(y) + 0x8000u) & 0xffff0000u;
  return hi | lo;
}

// ---------------- GEMM: C[M,N] = A[M,K](f32->bf16) @ W[N,K](f32->bf16)^T, f32 accum ----------------
#define TM 128
#define TN 64
#define TK 32
#define PAD 8   // pad LDS rows to 40 ushorts = 80B: 16B-aligned rows, 20-dword
                // bank stride (distinct banks for 16 lanes), enables b128 ds ops

template <int LVEC, bool GUARDED>
__device__ __forceinline__ void load_pack_A(const float* __restrict__ arow, int k0, int K,
                                            int ac, unsigned int (&areg)[8]) {
  #pragma unroll
  for (int i = 0; i < 16; i += LVEC) {
    int gk = k0 + ac + i;
    if (LVEC == 4) {                       // K % 4 == 0: rows 16B-aligned, gk%4==0
      float4 t = {0.f, 0.f, 0.f, 0.f};
      if (!GUARDED || gk < K) t = *(const float4*)(arow + gk);
      areg[i / 2]     = pack_bf16x2(t.x, t.y);
      areg[i / 2 + 1] = pack_bf16x2(t.z, t.w);
    } else {                               // K even: pair granularity, 8B-aligned
      float2 t = {0.f, 0.f};
      if (!GUARDED || gk < K) t = *(const float2*)(arow + gk);
      areg[i / 2] = pack_bf16x2(t.x, t.y);
    }
  }
}

template <int LVEC, bool GUARDED>
__device__ __forceinline__ void load_pack_B(const float* __restrict__ wrow, int k0, int K,
                                            int bc, unsigned int (&breg)[4]) {
  #pragma unroll
  for (int i = 0; i < 8; i += LVEC) {
    int gk = k0 + bc + i;
    if (LVEC == 4) {
      float4 t = {0.f, 0.f, 0.f, 0.f};
      if (!GUARDED || gk < K) t = *(const float4*)(wrow + gk);
      breg[i / 2]     = pack_bf16x2(t.x, t.y);
      breg[i / 2 + 1] = pack_bf16x2(t.z, t.w);
    } else {
      float2 t = {0.f, 0.f};
      if (!GUARDED || gk < K) t = *(const float2*)(wrow + gk);
      breg[i / 2] = pack_bf16x2(t.x, t.y);
    }
  }
}

__device__ __forceinline__ void store_tiles(unsigned short (*As)[TK + PAD],
                                            unsigned short (*Bs)[TK + PAD],
                                            int ar, int ac, int br, int bc,
                                            const unsigned int (&areg)[8],
                                            const unsigned int (&breg)[4]) {
  #pragma unroll
  for (int j = 0; j < 8; ++j)              // 32B contiguous, 16B-aligned -> 2x ds_store_b128
    *(unsigned int*)&As[ar][ac + 2 * j] = areg[j];
  #pragma unroll
  for (int j = 0; j < 4; ++j)              // 16B contiguous, 16B-aligned -> 1x ds_store_b128
    *(unsigned int*)&Bs[br][bc + 2 * j] = breg[j];
}

__device__ __forceinline__ void wmma_tiles(const unsigned short (*As)[TK + PAD],
                                           const unsigned short (*Bs)[TK + PAD],
                                           int wave, int lrow, int lhalf, floatx8* acc) {
  // A fragment (16x32 bf16): lane-half = K half, 32B contiguous -> ds_load_b128 x2
  Frag16 a;
  #pragma unroll
  for (int v = 0; v < 8; ++v) {
    int kk = lhalf * 16 + 2 * v;
    a.u[v] = *(const unsigned int*)&As[wave * 16 + lrow][kk];
  }
  // B fragments (32x16): lane = column, lane-half = K half; 32B contiguous -> b128 x2
  #pragma unroll
  for (int ct = 0; ct < 4; ++ct) {
    Frag16 b;
    #pragma unroll
    for (int v = 0; v < 8; ++v)
      b.u[v] = *(const unsigned int*)&Bs[ct * 16 + lrow][lhalf * 16 + 2 * v];
    acc[ct] = __builtin_amdgcn_wmma_f32_16x16x32_bf16(false, a.v, false, b.v,
                                                      (short)0, acc[ct], false, false);
  }
}

template <int LVEC>
__global__ __launch_bounds__(256)
void sage_wmma_gemm(const float* __restrict__ A, const float* __restrict__ W,
                    float* __restrict__ C, int N, int K)
{
  __shared__ unsigned short As[2][TM][TK + PAD];   // 2 x 10 KB
  __shared__ unsigned short Bs[2][TN][TK + PAD];   // 2 x 5 KB

  const int tid   = threadIdx.x;
  const int wave  = tid >> 5;      // 0..7 (wave32)
  const int lane  = tid & 31;
  const int lrow  = lane & 15;
  const int lhalf = lane >> 4;

  const int m0 = blockIdx.x * TM;
  const int n0 = blockIdx.y * TN;

  const int ar = tid >> 1;                  // 0..127
  const int ac = (tid & 1) * 16;            // 0 | 16
  const float* arow = A + (size_t)(m0 + ar) * K;

  const int br = tid >> 2;                  // 0..63
  const int bc = (tid & 3) * 8;             // 0,8,16,24
  const int wn = n0 + br;
  // clamp OOB weight rows to row 0: their products only land in accumulator
  // lanes whose column store is guarded by (col < N), so no branch needed.
  const float* wrow = W + (size_t)(wn < N ? wn : 0) * K;

  floatx8 acc[4] = {{}, {}, {}, {}};
  unsigned int areg[8], breg[4];

  const int K_main = K & ~(TK - 1);         // K >= 90 -> K_main >= 2 tiles always

  // prologue: stage tile 0 into buffer 0
  load_pack_A<LVEC, false>(arow, 0, K, ac, areg);
  load_pack_B<LVEC, false>(wrow, 0, K, bc, breg);
  store_tiles(As[0], Bs[0], ar, ac, br, bc, areg, breg);
  __syncthreads();

  // double-buffered main loop: one barrier per tile, loads overlap WMMA
  int cur = 0;
  for (int k0 = TK; k0 < K_main; k0 += TK) {
    __builtin_prefetch(arow + k0 + TK, 0, 3);
    load_pack_A<LVEC, false>(arow, k0, K, ac, areg);
    load_pack_B<LVEC, false>(wrow, k0, K, bc, breg);
    wmma_tiles(As[cur], Bs[cur], wave, lrow, lhalf, acc);
    store_tiles(As[cur ^ 1], Bs[cur ^ 1], ar, ac, br, bc, areg, breg);
    __syncthreads();
    cur ^= 1;
  }
  if (K_main < K) {                          // ragged tail tile (zero-padded)
    load_pack_A<LVEC, true>(arow, K_main, K, ac, areg);
    load_pack_B<LVEC, true>(wrow, K_main, K, bc, breg);
    wmma_tiles(As[cur], Bs[cur], wave, lrow, lhalf, acc);
    store_tiles(As[cur ^ 1], Bs[cur ^ 1], ar, ac, br, bc, areg, breg);
    __syncthreads();
    cur ^= 1;
  }
  wmma_tiles(As[cur], Bs[cur], wave, lrow, lhalf, acc);   // drain last buffer

  // C/D layout: lane = column, VGPR r -> row r + 8*lhalf
  const int crow = m0 + wave * 16 + lhalf * 8;
  #pragma unroll
  for (int ct = 0; ct < 4; ++ct) {
    int col = n0 + ct * 16 + lrow;
    if (col < N) {
      #pragma unroll
      for (int r = 0; r < 8; ++r) C[(size_t)(crow + r) * N + col] = acc[ct][r];
    }
  }
}

// ---------------- graph / elementwise kernels ----------------
__global__ void zero_u32_k(unsigned int* p, int n) {
  int i = blockIdx.x * 256 + threadIdx.x;
  if (i < n) p[i] = 0u;
}

__global__ void deg_count_k(const int* __restrict__ dst, int* __restrict__ deg) {
  int e = blockIdx.x * 256 + threadIdx.x;
  if (e < N_EDGES) atomicAdd(deg + dst[e], 1);
}

// grid = (N_EDGES, ceil(F/256)); edge index uniform per block -> scalarized
__global__ void edge_scatter_k(const float* __restrict__ H, const int* __restrict__ src,
                               const int* __restrict__ dst, float* __restrict__ acc, int F) {
  const int e = blockIdx.x;
  const int f = blockIdx.y * 256 + threadIdx.x;
  if (f >= F) return;
  atomicAdd(acc + (size_t)dst[e] * F + f, H[(size_t)src[e] * F + f]);
}

// grid = (N_NODES, ceil(F/256))
__global__ void sage_finalize_k(const float* __restrict__ acc, const float* __restrict__ Hr,
                                const float* __restrict__ bias, const int* __restrict__ deg,
                                float* __restrict__ Z, int F) {
  const int i = blockIdx.x;
  const int f = blockIdx.y * 256 + threadIdx.x;
  if (f >= F) return;
  const size_t idx = (size_t)i * F + f;
  float d = (float)deg[i];
  if (d < 1.f) d = 1.f;
  float z = acc[idx] / d + bias[f] + Hr[idx];
  Z[idx] = (z >= 0.f) ? z : LEAKY * z;
}

__global__ __launch_bounds__(256)
void bn_cols_k(const float* __restrict__ X, float* __restrict__ Y, int F) {
  const int f   = blockIdx.x;     // one block per feature column
  const int tid = threadIdx.x;
  float vals[16];
  float s = 0.f, s2 = 0.f;
  #pragma unroll
  for (int i = 0; i < 16; ++i) {
    float v = X[(size_t)(tid + i * 256) * F + f];
    vals[i] = v; s += v; s2 += v * v;
  }
  __shared__ float r0[256], r1[256];
  r0[tid] = s; r1[tid] = s2;
  __syncthreads();
  for (int o = 128; o > 0; o >>= 1) {
    if (tid < o) { r0[tid] += r0[tid + o]; r1[tid] += r1[tid + o]; }
    __syncthreads();
  }
  float mean = r0[0] * (1.f / (float)N_NODES);
  float var  = r1[0] * (1.f / (float)N_NODES) - mean * mean;
  float inv  = rsqrtf(var + BN_EPS);
  #pragma unroll
  for (int i = 0; i < 16; ++i)
    Y[(size_t)(tid + i * 256) * F + f] = (vals[i] - mean) * inv;
}

__global__ void pool_mlp_k(const float* __restrict__ X4,
                           const float* __restrict__ Wf1, const float* __restrict__ bf1,
                           const float* __restrict__ Wf2, const float* __restrict__ bf2,
                           const float* __restrict__ Wf3, const float* __restrict__ bf3,
                           float* __restrict__ out) {
  const int b = blockIdx.x;       // 16 graphs
  const int tid = threadIdx.x;    // 64 threads
  __shared__ float pooled[50], h1[32], h2[16];
  if (tid < 50) {
    float s = 0.f;
    for (int r = 0; r < N_NODES / 16; ++r)
      s += X4[(size_t)(b * (N_NODES / 16) + r) * 50 + tid];
    pooled[tid] = s;
  }
  __syncthreads();
  if (tid < 32) {
    float s = bf1[tid];
    for (int k = 0; k < 50; ++k) s += Wf1[tid * 50 + k] * pooled[k];
    h1[tid] = s;
  }
  __syncthreads();
  if (tid < 16) {
    float s = bf2[tid];
    for (int k = 0; k < 32; ++k) s += Wf2[tid * 32 + k] * h1[k];
    h2[tid] = s;
  }
  __syncthreads();
  if (tid == 0) {
    float s = bf3[0];
    for (int k = 0; k < 16; ++k) s += Wf3[k] * h2[k];
    out[b] = s;
  }
}

// ---------------- host orchestration ----------------
extern "C" void kernel_launch(void* const* d_in, const int* in_sizes, int n_in,
                              void* d_out, int out_size, void* d_ws, size_t ws_size,
                              hipStream_t stream) {
  (void)in_sizes; (void)n_in; (void)out_size; (void)ws_size;

  const float* x_in = (const float*)d_in[0];
  const int*   edge = (const int*)d_in[1];
  const int*   src  = edge;            // edge_index[0]
  const int*   dst  = edge + N_EDGES;  // edge_index[1]

  const float* Wl[4] = {(const float*)d_in[2], (const float*)d_in[5],
                        (const float*)d_in[8], (const float*)d_in[11]};
  const float* bl[4] = {(const float*)d_in[3], (const float*)d_in[6],
                        (const float*)d_in[9], (const float*)d_in[12]};
  const float* Wr[4] = {(const float*)d_in[4], (const float*)d_in[7],
                        (const float*)d_in[10], (const float*)d_in[13]};
  const float* Wf1 = (const float*)d_in[14]; const float* bf1 = (const float*)d_in[15];
  const float* Wf2 = (const float*)d_in[16]; const float* bf2 = (const float*)d_in[17];
  const float* Wf3 = (const float*)d_in[18]; const float* bf3 = (const float*)d_in[19];

  // workspace layout (~26.2 MB)
  const size_t NB_ELEMS = (size_t)N_NODES * 320;      // max feature width
  char* ws  = (char*)d_ws;
  int*   deg = (int*)ws;                              // 16 KB
  float* Hl  = (float*)(ws + (1 << 14));
  float* Hr  = Hl  + NB_ELEMS;
  float* acc = Hr  + NB_ELEMS;
  float* NB0 = acc + NB_ELEMS;
  float* NB1 = NB0 + NB_ELEMS;
  float* NB[2] = {NB0, NB1};

  // degree (recomputed every call; harness may poison ws)
  zero_u32_k<<<(N_NODES + 255) / 256, 256, 0, stream>>>((unsigned int*)deg, N_NODES);
  deg_count_k<<<N_EDGES / 256, 256, 0, stream>>>(dst, deg);

  const int Ks[4] = {8500, 320, 180, 90};
  const int Fs[4] = {320, 180, 90, 50};
  const float* Ain = x_in;

  for (int L = 0; L < 4; ++L) {
    const int K = Ks[L], F = Fs[L];
    dim3 gg(N_NODES / TM, (F + TN - 1) / TN);
    if ((K & 3) == 0) {   // 16B-aligned rows -> float4 staging loads
      sage_wmma_gemm<4><<<gg, 256, 0, stream>>>(Ain, Wl[L], Hl, F, K);
      sage_wmma_gemm<4><<<gg, 256, 0, stream>>>(Ain, Wr[L], Hr, F, K);
    } else {              // K=90: float2 staging loads
      sage_wmma_gemm<2><<<gg, 256, 0, stream>>>(Ain, Wl[L], Hl, F, K);
      sage_wmma_gemm<2><<<gg, 256, 0, stream>>>(Ain, Wr[L], Hr, F, K);
    }

    const int nf = N_NODES * F;
    const int fch = (F + 255) / 256;
    zero_u32_k<<<(nf + 255) / 256, 256, 0, stream>>>((unsigned int*)acc, nf);
    edge_scatter_k<<<dim3(N_EDGES, fch), 256, 0, stream>>>(Hl, src, dst, acc, F);
    sage_finalize_k<<<dim3(N_NODES, fch), 256, 0, stream>>>(acc, Hr, bl[L], deg, Hl, F);
    bn_cols_k<<<F, 256, 0, stream>>>(Hl, NB[L & 1], F);
    Ain = NB[L & 1];
  }

  pool_mlp_k<<<16, 64, 0, stream>>>(Ain, Wf1, bf1, Wf2, bf2, Wf3, bf3, (float*)d_out);
}
